// Mask2FormerDecoder_48670569398854
// MI455X (gfx1250) — compile-verified, matrix operand order checked by reading, not run on hardware
//
#include <hip/hip_runtime.h>

// ---------------- constants ----------------
#define LAYERS 9
#define DMODEL 256
#define NQ     100
#define NHEAD  8
#define HDIM   32
#define FFDIM  2048
#define NCLS   151
#define NB     8
#define HMASK  112
#define NPIX   (HMASK*HMASK)   // 12544

typedef __attribute__((ext_vector_type(16))) __bf16          v16bf;
typedef __attribute__((ext_vector_type(16))) unsigned short  v16us;
typedef __attribute__((ext_vector_type(8)))  unsigned short  v8us;
typedef __attribute__((ext_vector_type(8)))  float           v8f;

// fp32 -> bf16 round-to-nearest-even
__device__ __forceinline__ unsigned short f2bf(float x) {
  unsigned int u = __float_as_uint(x);
  u += 0x7FFFu + ((u >> 16) & 1u);
  return (unsigned short)(u >> 16);
}

// ------------------------------------------------------------------
// Batched GEMM on the WMMA pipe.    C[z] = scale*A[z]*B[z] + bias (+ReLU)
//   BT=1: B is NxK row-major (y = x @ W^T);  BT=0: B is KxN row-major.
// Batch z -> (zo=z/sub, zi=z%sub), base += zo*O + zi*I.
// 256 threads = 8 waves; block tile 64(M) x 64(N); K-step 32; LDS double
// buffered (one barrier per K-step). Each wave computes 32x16 = 2 WMMAs.
// Fragments staged pre-swizzled into the CDNA5 16-bit A/B VGPR layout:
// an aligned 8-wide K-block is contiguous in the fragment, so staging is
// two float4 global loads + one b128 LDS store per thread. The next
// K-step's addresses are prefetched (global_prefetch_b8).
// ------------------------------------------------------------------
template<int BT, int RELU>
__global__ __launch_bounds__(256)
void gemm_wmma(const float* __restrict__ A, const float* __restrict__ Bm,
               const float* __restrict__ bias, float* __restrict__ C,
               int M, int N, int K, int lda, int ldb, int ldc,
               int sub, long long aO, long long aI, long long bO, long long bI,
               long long cO, long long cI, float scale)
{
  __shared__ alignas(32) unsigned short smA[2][2048];  // 4 tiles x (32 lanes x 16)
  __shared__ alignas(32) unsigned short smB[2][2048];

  const int z  = blockIdx.z;
  const int zo = z / sub;
  const int zi = z - zo * sub;
  const float* Ab = A  + zo * aO + zi * aI;
  const float* Bb = Bm + zo * bO + zi * bI;
  float*       Cb = C  + zo * cO + zi * cI;

  const int rowBase = blockIdx.y * 64;
  const int colBase = blockIdx.x * 64;
  const int tid  = threadIdx.x;
  const int lane = tid & 31;
  const int wave = tid >> 5;
  const int mquad = wave >> 2;   // 0..1 -> M offset 32*mquad
  const int nidx  = wave & 3;    // 0..3 -> N offset 16*nidx

  // ---- A staging: thread owns one aligned 8-wide K-block of one row ----
  const int at   = tid >> 6;            // tile 0..3
  const int ar   = (tid >> 2) & 15;     // row in tile
  const int akb  = (tid & 3) * 8;       // k base within 32-step
  const int arow = rowBase + at * 16 + ar;
  const int aslot = at * 512 + (ar + ((akb >> 3) & 1) * 16) * 16 + (akb >> 4) * 8;

  // ---- B staging (BT): same shape as A with columns as rows ----
  const int btl  = tid >> 6;
  const int bc   = (tid >> 2) & 15;
  const int bkb  = (tid & 3) * 8;
  const int bcol = colBase + btl * 16 + bc;
  const int bslot = btl * 512 + (bc + ((bkb >> 3) & 1) * 16) * 16 + (bkb >> 4) * 8;

  // ---- B staging (BN): thread owns 8 consecutive columns of one k-row ----
  const int nt   = tid >> 6;
  const int bkk  = (tid >> 1) & 31;
  const int ncb  = (tid & 1) * 8;
  const int ncol0 = colBase + nt * 16 + ncb;
  const int nsbase = nt * 512 + ((bkk >> 3) & 1) * 256 + (bkk >> 4) * 8
                   + ((bkk & 7) >> 1) * 2 + (bkk & 1);

  auto stageA = [&](unsigned short* dst, int k0) {
    float v[8];
    const int kb = k0 + akb;
    if (arow < M && kb + 8 <= K) {
      const float4* p = (const float4*)(Ab + (long long)arow * lda + kb);
      float4 x = p[0], y = p[1];
      v[0]=x.x; v[1]=x.y; v[2]=x.z; v[3]=x.w; v[4]=y.x; v[5]=y.y; v[6]=y.z; v[7]=y.w;
      if (kb + 40 <= K) __builtin_prefetch(Ab + (long long)arow * lda + kb + 32, 0, 1);
    } else {
      #pragma unroll
      for (int j = 0; j < 8; ++j)
        v[j] = (arow < M && kb + j < K) ? Ab[(long long)arow * lda + kb + j] : 0.f;
    }
    v8us u;
    #pragma unroll
    for (int j = 0; j < 8; ++j) u[j] = f2bf(v[j]);
    *reinterpret_cast<v8us*>(dst + aslot) = u;
  };

  auto stageB = [&](unsigned short* dst, int k0) {
    if (BT) {
      float v[8];
      const int kb = k0 + bkb;
      if (bcol < N && kb + 8 <= K) {
        const float4* p = (const float4*)(Bb + (long long)bcol * ldb + kb);
        float4 x = p[0], y = p[1];
        v[0]=x.x; v[1]=x.y; v[2]=x.z; v[3]=x.w; v[4]=y.x; v[5]=y.y; v[6]=y.z; v[7]=y.w;
        if (kb + 40 <= K) __builtin_prefetch(Bb + (long long)bcol * ldb + kb + 32, 0, 1);
      } else {
        #pragma unroll
        for (int j = 0; j < 8; ++j)
          v[j] = (bcol < N && kb + j < K) ? Bb[(long long)bcol * ldb + kb + j] : 0.f;
      }
      v8us u;
      #pragma unroll
      for (int j = 0; j < 8; ++j) u[j] = f2bf(v[j]);
      *reinterpret_cast<v8us*>(dst + bslot) = u;
    } else {
      float v[8];
      const int kk = k0 + bkk;
      if (kk < K && ncol0 + 8 <= N) {
        const float4* p = (const float4*)(Bb + (long long)kk * ldb + ncol0);
        float4 x = p[0], y = p[1];
        v[0]=x.x; v[1]=x.y; v[2]=x.z; v[3]=x.w; v[4]=y.x; v[5]=y.y; v[6]=y.z; v[7]=y.w;
        if (kk + 32 < K) __builtin_prefetch(Bb + (long long)(kk + 32) * ldb + ncol0, 0, 1);
      } else {
        #pragma unroll
        for (int j = 0; j < 8; ++j)
          v[j] = (kk < K && ncol0 + j < N) ? Bb[(long long)kk * ldb + ncol0 + j] : 0.f;
      }
      #pragma unroll
      for (int j = 0; j < 8; ++j) dst[nsbase + (ncb + j) * 16] = f2bf(v[j]);
    }
  };

  v8f acc[2] = { {}, {} };
  const int nks = (K + 31) >> 5;
  stageA(smA[0], 0);
  stageB(smB[0], 0);
  int cur = 0;
  for (int it = 0; it < nks; ++it) {
    __syncthreads();
    union { v16us u; v16bf b; } fa0, fa1, fb;
    fa0.u = *reinterpret_cast<const v16us*>(&smA[cur][(mquad * 2 + 0) * 512 + lane * 16]);
    fa1.u = *reinterpret_cast<const v16us*>(&smA[cur][(mquad * 2 + 1) * 512 + lane * 16]);
    fb.u  = *reinterpret_cast<const v16us*>(&smB[cur][nidx * 512 + lane * 16]);
    if (it + 1 < nks) {          // stage next K-step into the other buffer
      stageA(smA[cur ^ 1], (it + 1) * 32);
      stageB(smB[cur ^ 1], (it + 1) * 32);
    }
    acc[0] = __builtin_amdgcn_wmma_f32_16x16x32_bf16(false, fa0.b, false, fb.b,
                                                     (short)0, acc[0], false, false);
    acc[1] = __builtin_amdgcn_wmma_f32_16x16x32_bf16(false, fa1.b, false, fb.b,
                                                     (short)0, acc[1], false, false);
    cur ^= 1;
  }

  // ---- epilogue: C/D layout, VGPR j -> M = j + 8*(lane>=16), N = lane&15 ----
  const int col = colBase + nidx * 16 + (lane & 15);
  float bval = 0.f;
  if (bias != nullptr && col < N) bval = bias[col];
  #pragma unroll
  for (int m = 0; m < 2; ++m) {
    const int rb = rowBase + (mquad * 2 + m) * 16 + (lane >> 4) * 8;
    #pragma unroll
    for (int j = 0; j < 8; ++j) {
      int row = rb + j;
      if (row < M && col < N) {
        float v = acc[m][j] * scale + bval;
        if (RELU) v = v > 0.f ? v : 0.f;
        Cb[(long long)row * ldc + col] = v;
      }
    }
  }
}

// ---------------- LayerNorm (optionally fused residual add) ----------------
__global__ __launch_bounds__(256)
void ln_fused(const float* __restrict__ x, const float* __restrict__ y,
              const float* __restrict__ g, const float* __restrict__ bta,
              float* __restrict__ out)
{
  __shared__ float red[256];
  int r = blockIdx.x, t = threadIdx.x;
  float v = x[(long long)r * DMODEL + t];
  if (y) v += y[(long long)r * DMODEL + t];
  red[t] = v; __syncthreads();
  for (int s = 128; s > 0; s >>= 1) { if (t < s) red[t] += red[t + s]; __syncthreads(); }
  float mean = red[0] * (1.f / DMODEL);
  __syncthreads();
  float d = v - mean;
  red[t] = d * d; __syncthreads();
  for (int s = 128; s > 0; s >>= 1) { if (t < s) red[t] += red[t + s]; __syncthreads(); }
  float var = red[0] * (1.f / DMODEL);
  out[(long long)r * DMODEL + t] = g[t] * d * rsqrtf(var + 1e-5f) + bta[t];
}

// ---------------- masked row softmax (in place) ----------------
__global__ __launch_bounds__(256)
void softmax_rows(float* __restrict__ s, const float* __restrict__ amask,
                  int Kk, int Q, int NHd)
{
  __shared__ float red[256];
  int row = blockIdx.x, t = threadIdx.x;      // row = (b*NH + h)*Q + q
  int q  = row % Q;
  int bh = row / Q;
  int b  = bh / NHd;
  float* p = s + (long long)row * Kk;
  const float* mr = amask ? amask + (long long)(b * Q + q) * Kk : nullptr;

  float mx = -3.0e38f;
  for (int k = t; k < Kk; k += 256) {
    float v = p[k] + (mr ? mr[k] : 0.f);
    mx = fmaxf(mx, v);
  }
  red[t] = mx; __syncthreads();
  for (int st = 128; st > 0; st >>= 1) { if (t < st) red[t] = fmaxf(red[t], red[t + st]); __syncthreads(); }
  mx = red[0]; __syncthreads();

  float sum = 0.f;
  for (int k = t; k < Kk; k += 256) {
    float v = __expf(p[k] + (mr ? mr[k] : 0.f) - mx);
    p[k] = v; sum += v;
  }
  red[t] = sum; __syncthreads();
  for (int st = 128; st > 0; st >>= 1) { if (t < st) red[t] += red[t + st]; __syncthreads(); }
  float inv = 1.f / red[0];
  for (int k = t; k < Kk; k += 256) p[k] *= inv;
}

// ------ attention mask: bilinear resize of prev mask logits, threshold<0,
//        unmask rows that are fully masked; emit additive float mask ------
__global__ __launch_bounds__(256)
void make_mask(const float* __restrict__ pm, float* __restrict__ am, int Hk, int Wk)
{
  __shared__ int red[256];
  int bq = blockIdx.x, t = threadIdx.x;        // bq = b*Q + q
  const float* src = pm + (long long)bq * NPIX;
  int Kk = Hk * Wk;
  float* dst = am + (long long)bq * Kk;
  float sh = (float)HMASK / Hk, sw = (float)HMASK / Wk;
  int allm = 1;
  for (int k = t; k < Kk; k += 256) {
    int kh = k / Wk, kw = k % Wk;
    float fy = (kh + 0.5f) * sh - 0.5f;
    float fx = (kw + 0.5f) * sw - 0.5f;
    int y0 = (int)floorf(fy), x0 = (int)floorf(fx);
    float wy = fy - y0, wx = fx - x0;
    int y0c = min(max(y0, 0), HMASK - 1), y1c = min(max(y0 + 1, 0), HMASK - 1);
    int x0c = min(max(x0, 0), HMASK - 1), x1c = min(max(x0 + 1, 0), HMASK - 1);
    float v00 = src[y0c * HMASK + x0c], v01 = src[y0c * HMASK + x1c];
    float v10 = src[y1c * HMASK + x0c], v11 = src[y1c * HMASK + x1c];
    float v = v00 * (1.f - wy) * (1.f - wx) + v01 * (1.f - wy) * wx
            + v10 * wy * (1.f - wx)        + v11 * wy * wx;
    int m = v < 0.f;
    dst[k] = m ? 1.f : 0.f;
    allm &= m;
  }
  red[t] = allm; __syncthreads();
  for (int st = 128; st > 0; st >>= 1) { if (t < st) red[t] &= red[t + st]; __syncthreads(); }
  int full = red[0];
  for (int k = t; k < Kk; k += 256) {
    int m = dst[k] > 0.5f;
    dst[k] = (m && !full) ? -1e30f : 0.f;
  }
}

// ---------------- in [B,D,K] -> out [B,K,D] ----------------
__global__ void transpose_mem(const float* __restrict__ in, float* __restrict__ out, int Kk)
{
  long long idx = (long long)blockIdx.x * blockDim.x + threadIdx.x;
  long long total = (long long)NB * Kk * DMODEL;
  if (idx >= total) return;
  int d = (int)(idx % DMODEL);
  long long r = idx / DMODEL;
  int k = (int)(r % Kk);
  int b = (int)(r / Kk);
  out[idx] = in[((long long)b * DMODEL + d) * Kk + k];
}

// ---------------- q init: broadcast(query_feat + query_embed) ----------------
__global__ void qinit(const float* __restrict__ qf, const float* __restrict__ qe,
                      float* __restrict__ q)
{
  int idx = blockIdx.x * blockDim.x + threadIdx.x;   // NB*NQ*DMODEL
  if (idx < NB * NQ * DMODEL) {
    int j = idx % (NQ * DMODEL);
    q[idx] = qf[j] + qe[j];
  }
}

// ---------------- host-side GEMM dispatcher ----------------
static inline void launch_gemm(int BT, int RELU,
    const float* A, const float* Bm, const float* bias, float* C,
    int M, int N, int K, int lda, int ldb, int ldc,
    int batches, int sub,
    long long aO, long long aI, long long bO, long long bI,
    long long cO, long long cI, float scale, hipStream_t st)
{
  dim3 g((N + 63) / 64, (M + 63) / 64, batches), b(256);
  if (BT) {
    if (RELU) gemm_wmma<1,1><<<g,b,0,st>>>(A,Bm,bias,C,M,N,K,lda,ldb,ldc,sub,aO,aI,bO,bI,cO,cI,scale);
    else      gemm_wmma<1,0><<<g,b,0,st>>>(A,Bm,bias,C,M,N,K,lda,ldb,ldc,sub,aO,aI,bO,bI,cO,cI,scale);
  } else {
    if (RELU) gemm_wmma<0,1><<<g,b,0,st>>>(A,Bm,bias,C,M,N,K,lda,ldb,ldc,sub,aO,aI,bO,bI,cO,cI,scale);
    else      gemm_wmma<0,0><<<g,b,0,st>>>(A,Bm,bias,C,M,N,K,lda,ldb,ldc,sub,aO,aI,bO,bI,cO,cI,scale);
  }
}

extern "C" void kernel_launch(void* const* d_in, const int* in_sizes, int n_in,
                              void* d_out, int out_size, void* d_ws, size_t ws_size,
                              hipStream_t stream)
{
  (void)in_sizes; (void)n_in; (void)out_size;
  // ----- inputs (setup_inputs order) -----
  const float* mask_features = (const float*)d_in[0];
  const float* mem[3] = { (const float*)d_in[1], (const float*)d_in[2], (const float*)d_in[3] };
  const float* query_feat  = (const float*)d_in[4];
  const float* query_embed = (const float*)d_in[5];
  const float* ca_in_w  = (const float*)d_in[6];
  const float* ca_in_b  = (const float*)d_in[7];
  const float* ca_out_w = (const float*)d_in[8];
  const float* ca_out_b = (const float*)d_in[9];
  const float* ca_ln_g  = (const float*)d_in[10];
  const float* ca_ln_b  = (const float*)d_in[11];
  const float* sa_in_w  = (const float*)d_in[12];
  const float* sa_in_b  = (const float*)d_in[13];
  const float* sa_out_w = (const float*)d_in[14];
  const float* sa_out_b = (const float*)d_in[15];
  const float* sa_ln_g  = (const float*)d_in[16];
  const float* sa_ln_b  = (const float*)d_in[17];
  const float* ffn_w1   = (const float*)d_in[18];
  const float* ffn_b1   = (const float*)d_in[19];
  const float* ffn_w2   = (const float*)d_in[20];
  const float* ffn_b2   = (const float*)d_in[21];
  const float* ffn_ln_g = (const float*)d_in[22];
  const float* ffn_ln_b = (const float*)d_in[23];
  const float* cls_w = (const float*)d_in[24];
  const float* cls_b = (const float*)d_in[25];
  const float* me_w1 = (const float*)d_in[26];
  const float* me_b1 = (const float*)d_in[27];
  const float* me_w2 = (const float*)d_in[28];
  const float* me_b2 = (const float*)d_in[29];
  const float* me_w3 = (const float*)d_in[30];
  const float* me_b3 = (const float*)d_in[31];
  const float* dn_g  = (const float*)d_in[32];
  const float* dn_b  = (const float*)d_in[33];

  // ----- output layout: pl_last | pm_last | stack_l[8] | stack_m[8] -----
  float* out = (float*)d_out;
  const size_t BQ = (size_t)NB * NQ;
  float* out_pl_last = out;
  float* out_pm_last = out + BQ * NCLS;
  float* out_stack_l = out_pm_last + BQ * NPIX;
  float* out_stack_m = out_stack_l + 8 * BQ * NCLS;

  // ----- workspace carve-up -----
  const int HKs[3] = { 14, 28, 56 };
  float* ws = (float*)d_ws;
  size_t off = 0;
  auto alloc = [&](size_t n) { float* p = ws + off; off += n; return p; };
  float* memfT[3];
  for (int m = 0; m < 3; ++m) memfT[m] = alloc((size_t)NB * HKs[m] * HKs[m] * DMODEL);
  const int KMAX = 56 * 56;
  float* q     = alloc(BQ * DMODEL);
  float* qp    = alloc(BQ * DMODEL);
  float* qkv   = alloc(BQ * 3 * DMODEL);              // packed self-attn q|k|v
  float* kv    = alloc((size_t)NB * KMAX * 2 * DMODEL); // packed cross-attn k|v
  float* sbuf  = alloc((size_t)NB * NHEAD * NQ * KMAX);
  float* amask = alloc((size_t)NB * NQ * KMAX);
  float* ctx   = alloc(BQ * DMODEL);
  float* xatt  = alloc(BQ * DMODEL);
  float* ffh   = alloc(BQ * FFDIM);
  float* nq    = alloc(BQ * DMODEL);
  float* meA   = alloc(BQ * DMODEL);
  float* meB   = alloc(BQ * DMODEL);
  // optional: mask_features transposed to [B, P, D] so the big mask einsum
  // uses the fully vectorized N x K ("BT") staging path
  float* mfT = nullptr;
  if ((off + (size_t)NB * NPIX * DMODEL) * sizeof(float) <= ws_size)
    mfT = alloc((size_t)NB * NPIX * DMODEL);

  const float ATT_SCALE = 0.17677669529663687f; // 1/sqrt(32)

  // ----- one-time prep -----
  for (int m = 0; m < 3; ++m) {
    int Kk = HKs[m] * HKs[m];
    long long total = (long long)NB * Kk * DMODEL;
    transpose_mem<<<(int)((total + 255) / 256), 256, 0, stream>>>(mem[m], memfT[m], Kk);
  }
  if (mfT) {
    long long total = (long long)NB * NPIX * DMODEL;
    transpose_mem<<<(int)((total + 255) / 256), 256, 0, stream>>>(mask_features, mfT, NPIX);
  }
  qinit<<<(NB * NQ * DMODEL + 255) / 256, 256, 0, stream>>>(query_feat, query_embed, q);

  for (int i = 0; i < LAYERS; ++i) {
    const int mi = i % 3;
    const int Hk = HKs[mi];
    const int Kk = Hk * Hk;
    const float* memT = memfT[mi];

    // ---- attention mask from previous layer's mask logits ----
    const float* am = nullptr;
    if (i > 0) {
      const float* prevpm = out_stack_m + (size_t)(i - 1) * BQ * NPIX;
      make_mask<<<(int)BQ, 256, 0, stream>>>(prevpm, amask, Hk, Hk);
      am = amask;
    }

    // ================= cross-attention =================
    const float* inw = ca_in_w + (size_t)i * 3 * DMODEL * DMODEL;
    const float* inb = ca_in_b + (size_t)i * 3 * DMODEL;
    // q projection (queries) and fused k|v projection (memory)
    launch_gemm(1,0, q,    inw,                inb,          qp,
                (int)BQ, DMODEL, DMODEL, DMODEL, DMODEL, DMODEL, 1,1, 0,0,0,0,0,0, 1.f, stream);
    launch_gemm(1,0, memT, inw + DMODEL*DMODEL, inb + DMODEL, kv,
                NB*Kk, 2*DMODEL, DMODEL, DMODEL, DMODEL, 2*DMODEL, 1,1, 0,0,0,0,0,0, 1.f, stream);
    // scores: per (b,h): [Q,32] x [Kk,32]^T
    launch_gemm(1,0, qp, kv, nullptr, sbuf,
                NQ, Kk, HDIM, DMODEL, 2*DMODEL, Kk, NB*NHEAD, NHEAD,
                (long long)NQ*DMODEL, HDIM, (long long)Kk*2*DMODEL, HDIM,
                (long long)NHEAD*NQ*Kk, (long long)NQ*Kk, ATT_SCALE, stream);
    softmax_rows<<<NB*NHEAD*NQ, 256, 0, stream>>>(sbuf, am, Kk, NQ, NHEAD);
    // context: per (b,h): [Q,Kk] x [Kk,32]   (v = kv columns 256..511)
    launch_gemm(0,0, sbuf, kv + DMODEL, nullptr, ctx,
                NQ, HDIM, Kk, Kk, 2*DMODEL, DMODEL, NB*NHEAD, NHEAD,
                (long long)NHEAD*NQ*Kk, (long long)NQ*Kk, (long long)Kk*2*DMODEL, HDIM,
                (long long)NQ*DMODEL, HDIM, 1.f, stream);
    launch_gemm(1,0, ctx, ca_out_w + (size_t)i*DMODEL*DMODEL, ca_out_b + (size_t)i*DMODEL, xatt,
                (int)BQ, DMODEL, DMODEL, DMODEL, DMODEL, DMODEL, 1,1, 0,0,0,0,0,0, 1.f, stream);
    ln_fused<<<(int)BQ, 256, 0, stream>>>(q, xatt, ca_ln_g + (size_t)i*DMODEL, ca_ln_b + (size_t)i*DMODEL, q);

    // ================= self-attention (fused qkv projection) =================
    const float* sinw = sa_in_w + (size_t)i * 3 * DMODEL * DMODEL;
    const float* sinb = sa_in_b + (size_t)i * 3 * DMODEL;
    launch_gemm(1,0, q, sinw, sinb, qkv,
                (int)BQ, 3*DMODEL, DMODEL, DMODEL, DMODEL, 3*DMODEL, 1,1, 0,0,0,0,0,0, 1.f, stream);
    launch_gemm(1,0, qkv, qkv + DMODEL, nullptr, sbuf,
                NQ, NQ, HDIM, 3*DMODEL, 3*DMODEL, NQ, NB*NHEAD, NHEAD,
                (long long)NQ*3*DMODEL, HDIM, (long long)NQ*3*DMODEL, HDIM,
                (long long)NHEAD*NQ*NQ, (long long)NQ*NQ, ATT_SCALE, stream);
    softmax_rows<<<NB*NHEAD*NQ, 256, 0, stream>>>(sbuf, nullptr, NQ, NQ, NHEAD);
    launch_gemm(0,0, sbuf, qkv + 2*DMODEL, nullptr, ctx,
                NQ, HDIM, NQ, NQ, 3*DMODEL, DMODEL, NB*NHEAD, NHEAD,
                (long long)NHEAD*NQ*NQ, (long long)NQ*NQ, (long long)NQ*3*DMODEL, HDIM,
                (long long)NQ*DMODEL, HDIM, 1.f, stream);
    launch_gemm(1,0, ctx, sa_out_w + (size_t)i*DMODEL*DMODEL, sa_out_b + (size_t)i*DMODEL, xatt,
                (int)BQ, DMODEL, DMODEL, DMODEL, DMODEL, DMODEL, 1,1, 0,0,0,0,0,0, 1.f, stream);
    ln_fused<<<(int)BQ, 256, 0, stream>>>(q, xatt, sa_ln_g + (size_t)i*DMODEL, sa_ln_b + (size_t)i*DMODEL, q);

    // ================= FFN =================
    launch_gemm(1,1, q,   ffn_w1 + (size_t)i*FFDIM*DMODEL, ffn_b1 + (size_t)i*FFDIM, ffh,
                (int)BQ, FFDIM, DMODEL, DMODEL, DMODEL, FFDIM, 1,1, 0,0,0,0,0,0, 1.f, stream);
    launch_gemm(1,0, ffh, ffn_w2 + (size_t)i*DMODEL*FFDIM, ffn_b2 + (size_t)i*DMODEL, xatt,
                (int)BQ, DMODEL, FFDIM, FFDIM, FFDIM, DMODEL, 1,1, 0,0,0,0,0,0, 1.f, stream);
    ln_fused<<<(int)BQ, 256, 0, stream>>>(q, xatt, ffn_ln_g + (size_t)i*DMODEL, ffn_ln_b + (size_t)i*DMODEL, q);

    // ================= heads =================
    ln_fused<<<(int)BQ, 256, 0, stream>>>(q, nullptr, dn_g, dn_b, nq);
    launch_gemm(1,1, nq,  me_w1, me_b1, meA, (int)BQ, DMODEL, DMODEL, DMODEL, DMODEL, DMODEL, 1,1, 0,0,0,0,0,0, 1.f, stream);
    launch_gemm(1,1, meA, me_w2, me_b2, meB, (int)BQ, DMODEL, DMODEL, DMODEL, DMODEL, DMODEL, 1,1, 0,0,0,0,0,0, 1.f, stream);
    launch_gemm(1,0, meB, me_w3, me_b3, meA, (int)BQ, DMODEL, DMODEL, DMODEL, DMODEL, DMODEL, 1,1, 0,0,0,0,0,0, 1.f, stream);
    // mask logits: per b, [Q,256] x [256, 12544]
    float* pmout = (i == LAYERS - 1) ? out_pm_last : out_stack_m + (size_t)i * BQ * NPIX;
    if (mfT) {
      launch_gemm(1,0, meA, mfT, nullptr, pmout,
                  NQ, NPIX, DMODEL, DMODEL, DMODEL, NPIX, NB, 1,
                  (long long)NQ*DMODEL, 0, (long long)NPIX*DMODEL, 0,
                  (long long)NQ*NPIX, 0, 1.f, stream);
    } else {
      launch_gemm(0,0, meA, mask_features, nullptr, pmout,
                  NQ, NPIX, DMODEL, DMODEL, NPIX, NPIX, NB, 1,
                  (long long)NQ*DMODEL, 0, (long long)DMODEL*NPIX, 0,
                  (long long)NQ*NPIX, 0, 1.f, stream);
    }
    // class logits
    float* plout = (i == LAYERS - 1) ? out_pl_last : out_stack_l + (size_t)i * BQ * NCLS;
    launch_gemm(1,0, nq, cls_w, cls_b, plout,
                (int)BQ, NCLS, DMODEL, DMODEL, DMODEL, NCLS, 1,1, 0,0,0,0,0,0, 1.f, stream);
  }
}
